// GCNConv_79139067396649
// MI455X (gfx1250) — matrix operand order repaired and checked by reading, not compiled, not run on hardware
//
#include <hip/hip_runtime.h>

// ---------- CDNA5 (gfx1250) types for WMMA ----------
typedef __attribute__((ext_vector_type(16))) __bf16 v16bf;
typedef __attribute__((ext_vector_type(2)))  __bf16 v2bf;
typedef __attribute__((ext_vector_type(8)))  float  v8f;

#define D_IN   256
#define D_OUT  64
#define BLOCK_ROWS 128     // 8 waves x 16 rows per 256-thread block

// f32 -> bf16, single value (used only in the tiny one-shot transpose kernel)
static __device__ __forceinline__ __bf16 f2bf(float f) {
    return (__bf16)f;   // clang: RNE conversion
}
// pack two f32 -> two bf16 in one dword (lo in [15:0], hi in [31:16])
// Type-level converts: lets clang select hw v_cvt_(pk_)bf16_f32 on gfx1250.
static __device__ __forceinline__ unsigned pk2bf(float lo, float hi) {
    v2bf r;
    r[0] = (__bf16)lo;
    r[1] = (__bf16)hi;
    return __builtin_bit_cast(unsigned, r);
}

// ============================================================================
// Kernel 0 (one-shot, tiny): Wt[n][k] = bf16(W[k][n]).  32 KB, L2-resident.
// ============================================================================
__global__ __launch_bounds__(256)
void transpose_w_bf16(const float* __restrict__ W, __bf16* __restrict__ Wt)
{
    int idx = blockIdx.x * 256 + threadIdx.x;     // 0 .. 16383
    if (idx < D_IN * D_OUT) {
        int k = idx >> 6;        // 0..255
        int n = idx & 63;        // 0..63
        Wt[(size_t)n * D_IN + k] = f2bf(W[idx]);  // coalesced read; 16K scattered b16 writes, negligible
    }
}

// ============================================================================
// Kernel 1: Xs[j,:] = degrees[j] * (X[j,:] @ W)      (BF16 WMMA, FP32 accum)
// Fully register-resident: no LDS, no barriers.
//  - A fragments loaded directly from X: per k-step the 2 lanes sharing a row
//    cover one contiguous 128B cacheline; every X byte read exactly once.
//  - B fragments: contiguous 32B v16bf loads from pre-transposed Wt (L2-hot).
// Each wave: 16 rows x 64 cols = 4 N-tiles x 8 K-steps = 32 v_wmma executed.
// unroll 2 keeps live fragments inside the 256-VGPR direct window
// (avoids s_set_vgpr_msb switching seen with full unroll).
// ============================================================================
__global__ __launch_bounds__(256)
void gcn_gemm_bf16_wmma(const float* __restrict__ X,
                        const __bf16* __restrict__ Wt,
                        const float* __restrict__ degrees,
                        float* __restrict__ Xs,
                        int n_nodes)
{
    const int tid     = threadIdx.x;
    const int lane    = tid & 31;
    const int wave    = tid >> 5;
    const int m       = lane & 15;
    const int hi      = lane >> 4;         // 0: lanes 0-15, 1: lanes 16-31
    const int rowbase = blockIdx.x * BLOCK_ROWS;

    int arow = rowbase + wave * 16 + m;    // this lane's A-matrix row
    if (arow >= n_nodes) arow = n_nodes - 1;   // clamp loads; stores guarded below
    const float* __restrict__ xrow = X + (size_t)arow * D_IN;

    // B fragment base: lane = N column (m), lanes 0-15 hold K 0..15,
    // lanes 16-31 hold K 16..31 of each 32-K step (2 K per VGPR).
    const __bf16* __restrict__ bbase = Wt + (size_t)m * D_IN + (hi ? 16 : 0);

    v8f c0 = {}, c1 = {}, c2 = {}, c3 = {};

#pragma unroll 2
    for (int ks = 0; ks < D_IN; ks += 32) {
        // ---- A fragment (16x32 bf16), contiguous K layout (matches B):
        //   lanes 0-15 : VGPR0-7 = K ks+0..15
        //   lanes 16-31: VGPR0-7 = K ks+16..31
        const int k0 = ks + (hi ? 16 : 0);
        const float4 a0 = *(const float4*)(xrow + k0);        // K k0..k0+3
        const float4 a1 = *(const float4*)(xrow + k0 + 4);    // K k0+4..k0+7
        const float4 a2 = *(const float4*)(xrow + k0 + 8);    // K k0+8..k0+11
        const float4 a3 = *(const float4*)(xrow + k0 + 12);   // K k0+12..k0+15
        union { v16bf v; unsigned u[8]; } af;
        af.u[0] = pk2bf(a0.x, a0.y);  af.u[1] = pk2bf(a0.z, a0.w);
        af.u[2] = pk2bf(a1.x, a1.y);  af.u[3] = pk2bf(a1.z, a1.w);
        af.u[4] = pk2bf(a2.x, a2.y);  af.u[5] = pk2bf(a2.z, a2.w);
        af.u[6] = pk2bf(a3.x, a3.y);  af.u[7] = pk2bf(a3.z, a3.w);

        // ---- B fragments: 32B contiguous per lane per N-tile (32B-aligned)
        const __bf16* bp = bbase + ks;
        v16bf b0 = *(const v16bf*)(bp);
        v16bf b1 = *(const v16bf*)(bp + 16 * D_IN);
        v16bf b2 = *(const v16bf*)(bp + 32 * D_IN);
        v16bf b3 = *(const v16bf*)(bp + 48 * D_IN);

        c0 = __builtin_amdgcn_wmma_f32_16x16x32_bf16(false, af.v, false, b0, (short)0, c0, false, false);
        c1 = __builtin_amdgcn_wmma_f32_16x16x32_bf16(false, af.v, false, b1, (short)0, c1, false, false);
        c2 = __builtin_amdgcn_wmma_f32_16x16x32_bf16(false, af.v, false, b2, (short)0, c2, false, false);
        c3 = __builtin_amdgcn_wmma_f32_16x16x32_bf16(false, af.v, false, b3, (short)0, c3, false, false);
    }

    // ---- epilogue: D layout (16x16 f32): lane 0-15 -> M=v, lane 16-31 -> M=v+8, N=m
    //      fuse per-source-row scale degrees[j] (d_j^{-1/2}) ----
    float dsc[8];
#pragma unroll
    for (int v = 0; v < 8; ++v) {
        int gr = rowbase + wave * 16 + (hi ? v + 8 : v);
        dsc[v] = (gr < n_nodes) ? degrees[gr] : 0.0f;
    }
#pragma unroll
    for (int v = 0; v < 8; ++v) {
        int gr = rowbase + wave * 16 + (hi ? v + 8 : v);
        if (gr < n_nodes) {
            float* o = Xs + (size_t)gr * D_OUT + m;
            o[0]  = c0[v] * dsc[v];
            o[16] = c1[v] * dsc[v];
            o[32] = c2[v] * dsc[v];
            o[48] = c3[v] * dsc[v];
        }
    }
}

// ============================================================================
// Kernel 2: out[i,:] = degrees[i] * sum_{e in row i} Xs[col[e], :]
// 16 lanes per node, float4 per lane => one coalesced 256B read per edge,
// served out of L2 (Xs is 25.6 MB << 192 MB L2). Software prefetch of next row.
// ============================================================================
__global__ __launch_bounds__(256)
void gcn_aggregate(const float* __restrict__ Xs,
                   const long long* __restrict__ rowptr,
                   const long long* __restrict__ colidx,
                   const float* __restrict__ degrees,
                   float* __restrict__ out,
                   int n_nodes)
{
    const int node = blockIdx.x * 16 + (threadIdx.x >> 4);
    const int l    = threadIdx.x & 15;
    if (node >= n_nodes) return;

    const long long s = rowptr[node];
    const long long e = rowptr[node + 1];

    float ax = 0.f, ay = 0.f, az = 0.f, aw = 0.f;
#pragma unroll 4
    for (long long p = s; p < e; ++p) {
        if (p + 1 < e) {
            long long jn = colidx[p + 1];
            __builtin_prefetch(Xs + (size_t)jn * D_OUT + l * 4, 0, 1); // global_prefetch_b8
        }
        long long j = colidx[p];
        const float4 v = *(const float4*)(Xs + (size_t)j * D_OUT + l * 4);
        ax += v.x; ay += v.y; az += v.z; aw += v.w;
    }
    const float d = degrees[node];
    float4 r; r.x = ax * d; r.y = ay * d; r.z = az * d; r.w = aw * d;
    *(float4*)(out + (size_t)node * D_OUT + l * 4) = r;
}

// ============================================================================
extern "C" void kernel_launch(void* const* d_in, const int* in_sizes, int n_in,
                              void* d_out, int out_size, void* d_ws, size_t ws_size,
                              hipStream_t stream)
{
    const float*     X       = (const float*)d_in[0];      // [N, 256] f32
    const float*     W       = (const float*)d_in[1];      // [256, 64] f32
    const long long* rowptr  = (const long long*)d_in[2];  // [N+1] i64
    const long long* colidx  = (const long long*)d_in[3];  // [E] i64
    const float*     degrees = (const float*)d_in[4];      // [N] f32
    float*           out     = (float*)d_out;              // [N, 64] f32

    const int n_nodes = in_sizes[4];

    // workspace layout: Xs [N*64 f32], then Wt [64*256 bf16] at 64B-aligned offset
    float* Xs = (float*)d_ws;
    size_t wt_off = ((size_t)n_nodes * D_OUT * sizeof(float) + 63) & ~(size_t)63;
    __bf16* Wt = (__bf16*)((char*)d_ws + wt_off);

    transpose_w_bf16<<<(D_IN * D_OUT + 255) / 256, 256, 0, stream>>>(W, Wt);

    const int gemm_blocks = (n_nodes + BLOCK_ROWS - 1) / BLOCK_ROWS;
    gcn_gemm_bf16_wmma<<<gemm_blocks, 256, 0, stream>>>(X, Wt, degrees, Xs, n_nodes);

    const int agg_blocks = (n_nodes + 15) / 16;
    gcn_aggregate<<<agg_blocks, 256, 0, stream>>>(Xs, rowptr, colidx, degrees, out, n_nodes);
}
